// BasicBlock_15762529976616
// MI455X (gfx1250) — 8-bit matrix operand order repaired and checked by reading, compiled, not run on hardware
//
#include <hip/hip_runtime.h>
#include <cstdint>

typedef __attribute__((ext_vector_type(8))) int v8i;

#define C_CH 256
#define HDIM 28
#define WDIM 28
#define NIMG 128
#define MTOT (NIMG * HDIM * WDIM)
#define KTOT (9 * C_CH)
#define KSTEPS (KTOT / 64)

#define BM 128
#define BN 128
#define LDA 80
#define LDB 80

#if defined(__gfx1250__) && __has_builtin(__builtin_amdgcn_global_load_async_to_lds_b128)
#define HAVE_ASYNC_LDS 1
typedef int b128_t __attribute__((vector_size(16)));
typedef __attribute__((address_space(1))) b128_t gb128;
typedef __attribute__((address_space(3))) b128_t lb128;
#else
#define HAVE_ASYNC_LDS 0
#endif

__device__ __forceinline__ void wait_async_le4() {
#if HAVE_ASYNC_LDS
#if __has_builtin(__builtin_amdgcn_s_wait_asynccnt)
    __builtin_amdgcn_s_wait_asynccnt(4);
#else
    asm volatile("s_wait_asynccnt 0x4" ::: "memory");
#endif
#endif
}
__device__ __forceinline__ void wait_async_le0() {
#if HAVE_ASYNC_LDS
#if __has_builtin(__builtin_amdgcn_s_wait_asynccnt)
    __builtin_amdgcn_s_wait_asynccnt(0);
#else
    asm volatile("s_wait_asynccnt 0x0" ::: "memory");
#endif
#endif
}

__global__ void zero_buf(int8_t* __restrict__ p, int n) {
    int i = (int)(blockIdx.x * blockDim.x + threadIdx.x);
    if (i < n) p[i] = 0;
}

__global__ void pack_input_nhwc(const float* __restrict__ x, int8_t* __restrict__ xb) {
    long i = (long)blockIdx.x * blockDim.x + threadIdx.x;
    if (i >= (long)MTOT * C_CH) return;
    int c = (int)(i & (C_CH - 1));
    long m = i >> 8;
    int w = (int)(m % WDIM);
    long t = m / WDIM;
    int h = (int)(t % HDIM);
    int n = (int)(t / HDIM);
    float v = x[(((long)n * C_CH + c) * HDIM + h) * WDIM + w];
    xb[i] = (v > 0.f) ? (int8_t)1 : ((v < 0.f) ? (int8_t)-1 : (int8_t)0);
}

__global__ void pack_weight(const float* __restrict__ w, int8_t* __restrict__ wbt) {
    long i = (long)blockIdx.x * blockDim.x + threadIdx.x;
    if (i >= (long)C_CH * KTOT) return;
    int k = (int)(i % KTOT);
    int o = (int)(i / KTOT);
    int ci = k & (C_CH - 1);
    int f  = k >> 8;
    int ky = f / 3, kx = f % 3;
    float v = w[(((long)o * C_CH + ci) * 3 + ky) * 3 + kx];
    wbt[i] = (v > 0.f) ? (int8_t)1 : ((v < 0.f) ? (int8_t)-1 : (int8_t)0);
}

__global__ __launch_bounds__(256) void bconv_gemm(
    const int8_t* __restrict__ ab,
    const int8_t* __restrict__ wbt,
    const int8_t* __restrict__ zpad,
    const float*  __restrict__ bias,
    const float*  __restrict__ resid,
    float* __restrict__ out)
{
    __shared__ alignas(16) int8_t ldsA[2][BM * LDA];
    __shared__ alignas(16) int8_t ldsB[2][BN * LDB];

    const int tid  = threadIdx.x;
    const int lane = tid & 31;
    const int wave = tid >> 5;
    const int wm   = wave & 3;
    const int wn   = wave >> 2;
    const int half = lane >> 4;
    const int ml   = lane & 15;
    const int nl   = lane & 15;

    const int m0 = blockIdx.y * BM;
    const int n0 = blockIdx.x * BN;

    const int r0  = tid >> 2;
    const int r1  = r0 + 64;
    const int off = (tid & 3) * 16;

    const int mA0 = m0 + r0, mA1 = m0 + r1;
    const int w0 = mA0 % WDIM, h0 = (mA0 / WDIM) % HDIM, img0 = mA0 / (WDIM * HDIM);
    const int w1p = mA1 % WDIM, h1 = (mA1 / WDIM) % HDIM, img1 = mA1 / (WDIM * HDIM);
    const int8_t* baseA0 = ab + ((((long)img0 * HDIM + h0) * WDIM + w0) << 8) + off;
    const int8_t* baseA1 = ab + ((((long)img1 * HDIM + h1) * WDIM + w1p) << 8) + off;
    const int8_t* zp     = zpad + off;
    const int8_t* baseB0 = wbt + (long)(n0 + r0) * KTOT + off;
    const int8_t* baseB1 = wbt + (long)(n0 + r1) * KTOT + off;

    auto stage = [&](int ks, int bi) {
        const int f  = ks >> 2;
        const int c0 = (ks & 3) * 64;
        const int ky = f / 3, kx = f % 3;
        const int fd = ((ky - 1) * WDIM + (kx - 1)) << 8;

        const int hi0 = h0 + ky - 1, wi0 = w0 + kx - 1;
        const int hi1 = h1 + ky - 1, wi1 = w1p + kx - 1;
        const bool ok0 = (hi0 >= 0) & (hi0 < HDIM) & (wi0 >= 0) & (wi0 < WDIM);
        const bool ok1 = (hi1 >= 0) & (hi1 < HDIM) & (wi1 >= 0) & (wi1 < WDIM);
        const int8_t* gA0 = ok0 ? (baseA0 + fd + c0) : zp;
        const int8_t* gA1 = ok1 ? (baseA1 + fd + c0) : zp;
        const int8_t* gB0 = baseB0 + ks * 64;
        const int8_t* gB1 = baseB1 + ks * 64;
        int8_t* lA0 = &ldsA[bi][r0 * LDA + off];
        int8_t* lA1 = &ldsA[bi][r1 * LDA + off];
        int8_t* lB0 = &ldsB[bi][r0 * LDB + off];
        int8_t* lB1 = &ldsB[bi][r1 * LDB + off];
#if HAVE_ASYNC_LDS
        __builtin_amdgcn_global_load_async_to_lds_b128((gb128*)gA0, (lb128*)lA0, 0, 0);
        __builtin_amdgcn_global_load_async_to_lds_b128((gb128*)gA1, (lb128*)lA1, 0, 0);
        __builtin_amdgcn_global_load_async_to_lds_b128((gb128*)gB0, (lb128*)lB0, 0, 0);
        __builtin_amdgcn_global_load_async_to_lds_b128((gb128*)gB1, (lb128*)lB1, 0, 0);
#else
        *(int4*)lA0 = *(const int4*)gA0;
        *(int4*)lA1 = *(const int4*)gA1;
        *(int4*)lB0 = *(const int4*)gB0;
        *(int4*)lB1 = *(const int4*)gB1;
#endif
    };

    v8i acc[2][4];
    const v8i vzero = {0, 0, 0, 0, 0, 0, 0, 0};
#pragma unroll
    for (int tm = 0; tm < 2; ++tm)
#pragma unroll
        for (int tn = 0; tn < 4; ++tn) acc[tm][tn] = vzero;

    stage(0, 0);

    for (int ks = 0; ks < KSTEPS; ++ks) {
        const int bi = ks & 1;
        if (ks + 1 < KSTEPS) {
            stage(ks + 1, bi ^ 1);
            wait_async_le4();
        } else {
            wait_async_le0();
        }
        __syncthreads();

        v8i af[2], bf[4];
#pragma unroll
        for (int tm = 0; tm < 2; ++tm) {
            const int8_t* ar = &ldsA[bi][(wm * 32 + tm * 16 + ml) * LDA];
            v8i a;
#pragma unroll
            for (int j = 0; j < 8; ++j) {
                int kb = (j >> 1) * 16 + half * 8 + (j & 1) * 4;
                a[j] = *(const int*)(ar + kb);
            }
            af[tm] = a;
        }
#pragma unroll
        for (int tn = 0; tn < 4; ++tn) {
            const int8_t* br = &ldsB[bi][(wn * 64 + tn * 16 + nl) * LDB];
            v8i b;
#pragma unroll
            for (int j = 0; j < 8; ++j) {
                int kb = (j >> 1) * 16 + half * 8 + (j & 1) * 4;
                b[j] = *(const int*)(br + kb);
            }
            bf[tn] = b;
        }

#pragma unroll
        for (int tm = 0; tm < 2; ++tm)
#pragma unroll
            for (int tn = 0; tn < 4; ++tn)
                acc[tm][tn] = __builtin_amdgcn_wmma_i32_16x16x64_iu8(
                    true, af[tm], true, bf[tn], acc[tm][tn], false, false);

        __syncthreads();
    }

#pragma unroll
    for (int tm = 0; tm < 2; ++tm) {
#pragma unroll
        for (int r = 0; r < 8; ++r) {
            const int m_g = m0 + wm * 32 + tm * 16 + r + 8 * half;
            long rbase = 0;
            if (resid) {
                const int ww = m_g % WDIM;
                const int hh = (m_g / WDIM) % HDIM;
                const int nn = m_g / (WDIM * HDIM);
                rbase = ((long)nn * C_CH) * (HDIM * WDIM) + hh * WDIM + ww;
            }
            float* orow = out + (long)m_g * C_CH;
#pragma unroll
            for (int tn = 0; tn < 4; ++tn) {
                const int n_g = n0 + wn * 64 + tn * 16 + nl;
                float v = (float)acc[tm][tn][r] + bias[n_g];
                if (resid) v += resid[rbase + (long)n_g * (HDIM * WDIM)];
                orow[n_g] = v;
            }
        }
    }
}

__global__ __launch_bounds__(256) void channel_stats(const float* __restrict__ raw,
                                                     float* __restrict__ mean,
                                                     float* __restrict__ invstd) {
    __shared__ float ssum[256], ssq[256];
    const int c = blockIdx.x;
    const int tid = threadIdx.x;
    float s = 0.f, q = 0.f;
    for (int m = tid; m < MTOT; m += 256) {
        float v = raw[(long)m * C_CH + c];
        s += v;
        q += v * v;
    }
    ssum[tid] = s; ssq[tid] = q;
    __syncthreads();
    for (int st = 128; st > 0; st >>= 1) {
        if (tid < st) { ssum[tid] += ssum[tid + st]; ssq[tid] += ssq[tid + st]; }
        __syncthreads();
    }
    if (tid == 0) {
        float mu  = ssum[0] / (float)MTOT;
        float var = ssq[0] / (float)MTOT - mu * mu;
        mean[c]   = mu;
        invstd[c] = rsqrtf(var + 1e-5f);
    }
}

__global__ void bn_sign(const float* __restrict__ raw, const float* __restrict__ mean,
                        const float* __restrict__ inv, const float* __restrict__ gamma,
                        const float* __restrict__ beta, int8_t* __restrict__ actb) {
    long i = (long)blockIdx.x * blockDim.x + threadIdx.x;
    if (i >= (long)MTOT * C_CH) return;
    int c = (int)(i & (C_CH - 1));
    float y = (raw[i] - mean[c]) * inv[c] * gamma[c] + beta[c];
    actb[i] = (y > 0.f) ? (int8_t)1 : ((y < 0.f) ? (int8_t)-1 : (int8_t)0);
}

__global__ void bn_tanh_nchw(const float* __restrict__ raw, const float* __restrict__ mean,
                             const float* __restrict__ inv, const float* __restrict__ gamma,
                             const float* __restrict__ beta, float* __restrict__ out) {
    long i = (long)blockIdx.x * blockDim.x + threadIdx.x;
    if (i >= (long)MTOT * C_CH) return;
    int c = (int)(i & (C_CH - 1));
    long m = i >> 8;
    float y = (raw[i] - mean[c]) * inv[c] * gamma[c] + beta[c];
    y = fminf(1.f, fmaxf(-1.f, y));
    int w = (int)(m % WDIM);
    long t = m / WDIM;
    int h = (int)(t % HDIM);
    int n = (int)(t / HDIM);
    out[(((long)n * C_CH + c) * HDIM + h) * WDIM + w] = y;
}

extern "C" void kernel_launch(void* const* d_in, const int* in_sizes, int n_in,
                              void* d_out, int out_size, void* d_ws, size_t ws_size,
                              hipStream_t stream) {
    const float* x   = (const float*)d_in[0];
    const float* w1  = (const float*)d_in[1];
    const float* b1  = (const float*)d_in[2];
    const float* g1  = (const float*)d_in[3];
    const float* be1 = (const float*)d_in[4];
    const float* w2  = (const float*)d_in[5];
    const float* b2  = (const float*)d_in[6];
    const float* g2  = (const float*)d_in[7];
    const float* be2 = (const float*)d_in[8];
    float* out = (float*)d_out;

    const size_t MC = (size_t)MTOT * C_CH;
    uint8_t* ws = (uint8_t*)d_ws;
    size_t off = 0;
    auto carve = [&](size_t bytes) -> void* {
        void* p = ws + off;
        off += (bytes + 255) & ~(size_t)255;
        return p;
    };
    int8_t* actb = (int8_t*)carve(MC);
    float*  raw  = (float*)carve(MC * sizeof(float));
    int8_t* wbt1 = (int8_t*)carve((size_t)C_CH * KTOT);
    int8_t* wbt2 = (int8_t*)carve((size_t)C_CH * KTOT);
    int8_t* zpad = (int8_t*)carve(256);
    float* mean1 = (float*)carve(C_CH * sizeof(float));
    float* inv1  = (float*)carve(C_CH * sizeof(float));
    float* mean2 = (float*)carve(C_CH * sizeof(float));
    float* inv2  = (float*)carve(C_CH * sizeof(float));

    const int ew_blocks = (int)((MC + 255) / 256);
    const int wk_blocks = (int)(((size_t)C_CH * KTOT + 255) / 256);
    dim3 gemm_grid(C_CH / BN, MTOT / BM);

    zero_buf<<<1, 256, 0, stream>>>(zpad, 256);
    pack_input_nhwc<<<ew_blocks, 256, 0, stream>>>(x, actb);
    pack_weight<<<wk_blocks, 256, 0, stream>>>(w1, wbt1);
    pack_weight<<<wk_blocks, 256, 0, stream>>>(w2, wbt2);

    bconv_gemm<<<gemm_grid, 256, 0, stream>>>(actb, wbt1, zpad, b1, nullptr, raw);
    channel_stats<<<C_CH, 256, 0, stream>>>(raw, mean1, inv1);
    bn_sign<<<ew_blocks, 256, 0, stream>>>(raw, mean1, inv1, g1, be1, actb);

    bconv_gemm<<<gemm_grid, 256, 0, stream>>>(actb, wbt2, zpad, b2, x, raw);
    channel_stats<<<C_CH, 256, 0, stream>>>(raw, mean2, inv2);
    bn_tanh_nchw<<<ew_blocks, 256, 0, stream>>>(raw, mean2, inv2, g2, be2, out);
}
